// AttentionFusion_19052474925328
// MI455X (gfx1250) — matrix operand order repaired and checked by reading, not compiled, not run on hardware
//
#include <hip/hip_runtime.h>
#include <hip/hip_bf16.h>

typedef __attribute__((ext_vector_type(16))) __bf16        v16bf;
typedef __attribute__((ext_vector_type(8)))  float         v8f;
typedef __attribute__((ext_vector_type(8)))  unsigned int  v8u;

#define EPSF 1e-5f

__device__ __forceinline__ unsigned short f2bf(float f) {
  __bf16 h = (__bf16)f;                        // RNE, native cvt where available
  return __builtin_bit_cast(unsigned short, h);
}
__device__ __forceinline__ float bf2f(unsigned short h) {
  return __uint_as_float(((unsigned int)h) << 16);
}
__device__ __forceinline__ float bflo(unsigned int w) { return __uint_as_float(w << 16); }
__device__ __forceinline__ float bfhi(unsigned int w) { return __uint_as_float(w & 0xffff0000u); }

// ---------------------------------------------------------------------------
// A fragment (16x32 bf16) from row-major bf16 LDS tile, leading dim ld (mult of 8).
// CDNA5 layout: lane m=lane%16; VGPR v packs K pair at
// K = (v>=4?16:0) + (lane>=16?8:0) + (v%4)*2  -> two contiguous 16B chunks.
__device__ __forceinline__ v16bf lds_a_frag(const unsigned short* p, int row0, int k0, int ld) {
  int lane = threadIdx.x & 31;
  int m = row0 + (lane & 15);
  int kbase = k0 + ((lane >> 4) << 3);
  const uint4* b = (const uint4*)(p + m * ld + kbase);   // 16B aligned
  uint4 lo = b[0];        // K pairs {0,2,4,6}
  uint4 hi = b[2];        // K pairs {16,18,20,22} (+32 bytes)
  v8u u;
  u[0] = lo.x; u[1] = lo.y; u[2] = lo.z; u[3] = lo.w;
  u[4] = hi.x; u[5] = hi.y; u[6] = hi.z; u[7] = hi.w;
  return __builtin_bit_cast(v16bf, u);
}

// B fragment (32x16 bf16) from pre-packed register-image weights:
// tile = 256 dwords, lane-major (lane*8 dwords contiguous) -> 2x global_load_b128.
__device__ __forceinline__ v16bf glb_b_frag_pk(const unsigned int* __restrict__ Wp, int tileIdx) {
  int lane = threadIdx.x & 31;
  const uint4* p = (const uint4*)(Wp + (size_t)tileIdx * 256 + lane * 8);
  uint4 a = p[0], b = p[1];
  v8u u;
  u[0] = a.x; u[1] = a.y; u[2] = a.z; u[3] = a.w;
  u[4] = b.x; u[5] = b.y; u[6] = b.z; u[7] = b.w;
  return __builtin_bit_cast(v16bf, u);
}

// C fragment (16x16 f32): lane n=lane%16, VGPR r holds M = r + 8*(lane/16).
__device__ __forceinline__ void store_c_bf16(unsigned short* p, v8f c, int row0, int n0, int ld) {
  int lane = threadIdx.x & 31;
  int n = n0 + (lane & 15);
  int mbase = row0 + ((lane >> 4) << 3);
#pragma unroll
  for (int r = 0; r < 8; ++r) p[(mbase + r) * ld + n] = f2bf(c[r]);
}

// ---------------------------------------------------------------- pack kernel
// Convert row-major f32 W[K][N] into bf16 B-fragment register images:
// out[tile][lane][v] dword = {bf16(W[k+1][n]), bf16(W[k][n])},
// tile = kt*(N/16)+nt, k = kt*32 + (v>=4?16:0) + (lane/16)*8 + 2*(v%4), n = nt*16 + lane%16.
__global__ void __launch_bounds__(256) pack_b_kernel(const float* __restrict__ W,
                                                     unsigned int* __restrict__ out,
                                                     int K, int N) {
  int i = blockIdx.x * 256 + threadIdx.x;          // dword index
  int total = (K * N) >> 1;
  if (i >= total) return;
  int ntiles = N >> 4;
  int tile = i >> 8;
  int r = i & 255;
  int lane = r >> 3;
  int v = r & 7;
  int kt = tile / ntiles;
  int nt = tile - kt * ntiles;
  int k = kt * 32 + ((v >> 2) << 4) + ((lane >> 4) << 3) + 2 * (v & 3);
  int n = nt * 16 + (lane & 15);
  unsigned int lo = f2bf(W[(size_t)k * N + n]);
  unsigned int hi = f2bf(W[(size_t)(k + 1) * N + n]);
  out[i] = lo | (hi << 16);
}

// ---------------------------------------------------------------- init + scatter
__global__ void __launch_bounds__(256) init_ctx_kernel(float4* __restrict__ ctx4, long long n4) {
  long long i = (long long)blockIdx.x * 256 + threadIdx.x;
  if (i < n4) ctx4[i] = make_float4(1.f, 1.f, 1.f, 1.f);
}

__global__ void __launch_bounds__(256) scatter_kernel(const float* __restrict__ img,
                                                      const int* __restrict__ inds2d,
                                                      const int* __restrict__ inds3d,
                                                      float* __restrict__ ctx, int M) {
  int wid = (int)((blockIdx.x * 256 + threadIdx.x) >> 5);
  int lane = threadIdx.x & 31;
  if (wid >= M) return;
  int x = inds2d[2 * wid + 0];
  int y = inds2d[2 * wid + 1];
  int n = inds3d[2 * wid + 0];
  int k = inds3d[2 * wid + 1];
  const float4* src = (const float4*)(img + ((size_t)y * 640 + (size_t)x) * 128);
  float4* dst = (float4*)(ctx + ((size_t)n * 64 + (size_t)k) * 128);
  dst[lane] = src[lane];
}

// ---------------------------------------------------------------- attention
// one block (8 waves) per point: LN, q/k/v, attention, Wo, residual.
__global__ void __launch_bounds__(256) attn_kernel(
    const float* __restrict__ pf, const float* __restrict__ ctx,
    const float* __restrict__ lnqg, const float* __restrict__ lnqb,
    const float* __restrict__ lncg, const float* __restrict__ lncb,
    const float* __restrict__ Wq, const unsigned int* __restrict__ Wkp,
    const unsigned int* __restrict__ Wvp, const float* __restrict__ Wo,
    const float* __restrict__ bo, float* __restrict__ xres) {
  __shared__ __align__(16) float ctxf[64 * 128];           // 32KB f32 ctx, reused for k/v bf16
  __shared__ __align__(16) unsigned short cnb[64 * 128];   // 16KB layernormed ctx bf16
  __shared__ float xs[256];
  __shared__ float xn[256];
  __shared__ float qv[128];
  __shared__ float sc[64];
  __shared__ float outv[128];
  __shared__ float wred[16];
  __shared__ float red[2];
  unsigned short* kvb = (unsigned short*)ctxf;             // [2][64][128] bf16 (k then v)

  const int tid = threadIdx.x;
  const int lane = tid & 31;
  const int wave = tid >> 5;
  const int n = blockIdx.x;

  // coalesced load of ctx tile (64x128 f32)
  {
    const float4* src = (const float4*)(ctx + (size_t)n * 8192);
    float4* dst = (float4*)ctxf;
#pragma unroll
    for (int i = tid; i < 2048; i += 256) dst[i] = src[i];
  }
  // x + LN_q (256 values, block reduce)
  float xv = pf[(size_t)n * 256 + tid];
  xs[tid] = xv;
  float s = xv, s2 = xv * xv;
#pragma unroll
  for (int m = 1; m < 32; m <<= 1) { s += __shfl_xor(s, m, 32); s2 += __shfl_xor(s2, m, 32); }
  if (lane == 0) { wred[wave] = s; wred[8 + wave] = s2; }
  __syncthreads();
  if (tid == 0) {
    float ts = 0.f, ts2 = 0.f;
    for (int w = 0; w < 8; ++w) { ts += wred[w]; ts2 += wred[8 + w]; }
    float mu = ts * (1.f / 256.f);
    float var = ts2 * (1.f / 256.f) - mu * mu;
    red[0] = mu; red[1] = rsqrtf(var + EPSF);
  }
  __syncthreads();
  xn[tid] = (xs[tid] - red[0]) * red[1] * lnqg[tid] + lnqb[tid];

  // LN of ctx rows -> cnb bf16 (4 threads per row)
  {
    int row = tid >> 2, l4 = tid & 3;
    float rs1 = 0.f, rs2 = 0.f;
#pragma unroll
    for (int j = 0; j < 32; ++j) {
      float f = ctxf[row * 128 + l4 + 4 * j];
      rs1 += f; rs2 += f * f;
    }
    rs1 += __shfl_xor(rs1, 1, 32); rs1 += __shfl_xor(rs1, 2, 32);
    rs2 += __shfl_xor(rs2, 1, 32); rs2 += __shfl_xor(rs2, 2, 32);
    float mu = rs1 * (1.f / 128.f);
    float var = rs2 * (1.f / 128.f) - mu * mu;
    float ir = rsqrtf(var + EPSF);
#pragma unroll
    for (int j = 0; j < 32; ++j) {
      int c = l4 + 4 * j;
      float f = ctxf[row * 128 + c];
      cnb[row * 128 + c] = f2bf((f - mu) * ir * lncg[c] + lncb[c]);
    }
  }
  __syncthreads();   // cnb + xn ready; ctxf free for reuse

  // q = xn @ Wq (1x256 @ 256x128), coalesced columns
  if (tid < 128) {
    float acc = 0.f;
    for (int c = 0; c < 256; ++c) acc = fmaf(xn[c], Wq[c * 128 + tid], acc);
    qv[tid] = acc;
  }

  // k = cn@Wk, v = cn@Wv (64x128 each) via bf16 WMMA; 64 output tiles over 8 waves
  for (int t = wave; t < 64; t += 8) {
    int which = t >> 5;          // 0 -> k, 1 -> v
    int tt = t & 31;
    int mt = tt >> 3;            // 0..3
    int nt = tt & 7;             // 0..7
    const unsigned int* Wp = which ? Wvp : Wkp;
    v8f acc = {};
#pragma unroll
    for (int kt = 0; kt < 4; ++kt) {
      v16bf a = lds_a_frag(cnb, mt * 16, kt * 32, 128);
      v16bf b = glb_b_frag_pk(Wp, kt * 8 + nt);
      acc = __builtin_amdgcn_wmma_f32_16x16x32_bf16(false, a, false, b, (short)0, acc, false, false);
    }
    store_c_bf16(kvb + which * 8192, acc, mt * 16, nt * 16, 128);
  }
  __syncthreads();

  // scores = q . k_r * SCALE (uint4 reads: 8 bf16 per 16B)
  if (tid < 64) {
    const uint4* krow = (const uint4*)(kvb + tid * 128);
    float acc = 0.f;
#pragma unroll
    for (int c8 = 0; c8 < 16; ++c8) {
      uint4 w = krow[c8];
      const float* qp = qv + c8 * 8;
      acc = fmaf(qp[0], bflo(w.x), acc);
      acc = fmaf(qp[1], bfhi(w.x), acc);
      acc = fmaf(qp[2], bflo(w.y), acc);
      acc = fmaf(qp[3], bfhi(w.y), acc);
      acc = fmaf(qp[4], bflo(w.z), acc);
      acc = fmaf(qp[5], bfhi(w.z), acc);
      acc = fmaf(qp[6], bflo(w.w), acc);
      acc = fmaf(qp[7], bfhi(w.w), acc);
    }
    sc[tid] = acc * 0.08838834764831845f;   // 128^-0.5
  }
  __syncthreads();
  if (tid == 0) {
    float mx = sc[0];
    for (int r = 1; r < 64; ++r) mx = fmaxf(mx, sc[r]);
    red[0] = mx;
  }
  __syncthreads();
  if (tid < 64) sc[tid] = __expf(sc[tid] - red[0]);
  __syncthreads();
  if (tid == 0) {
    float sm = 0.f;
    for (int r = 0; r < 64; ++r) sm += sc[r];
    red[1] = 1.f / sm;
  }
  __syncthreads();
  // out = softmax(scores) @ v  (1x64 @ 64x128); 2 columns per thread (full dword use)
  if (tid < 64) {
    const unsigned int* vb = (const unsigned int*)(kvb + 8192) + tid;  // cols 2t,2t+1
    float a0 = 0.f, a1 = 0.f;
#pragma unroll 4
    for (int r = 0; r < 64; ++r) {
      unsigned int w = vb[r * 64];
      float sr = sc[r];
      a0 = fmaf(sr, bflo(w), a0);
      a1 = fmaf(sr, bfhi(w), a1);
    }
    outv[2 * tid]     = a0 * red[1];
    outv[2 * tid + 1] = a1 * red[1];
  }
  __syncthreads();
  // y = out @ Wo + bo + x  -> residual buffer
  {
    float acc = 0.f;
    for (int c = 0; c < 128; ++c) acc = fmaf(outv[c], Wo[c * 256 + tid], acc);
    xres[(size_t)n * 256 + tid] = acc + bo[tid] + xs[tid];
  }
}

// ---------------------------------------------------------------- FFN
// one block per 16 points: LN -> GEMM1(+bias,gate) -> GEMM2(+bias,res,relu)
__global__ void __launch_bounds__(256) ffn_kernel(
    const float* __restrict__ xres, const float* __restrict__ lnfg,
    const float* __restrict__ lnfb, const unsigned int* __restrict__ W1p,
    const float* __restrict__ b1, const unsigned int* __restrict__ W2p,
    const float* __restrict__ b2, float* __restrict__ out) {
  __shared__ __align__(16) float xt[16 * 256];            // 16KB residual rows
  __shared__ __align__(16) unsigned short a1[16 * 256];   // 8KB  LN rows bf16
  __shared__ __align__(16) unsigned short gt[16 * 1024];  // 32KB gated hidden bf16

  const int tid = threadIdx.x;
  const int lane = tid & 31;
  const int wave = tid >> 5;
  const int row0 = blockIdx.x * 16;

  {
    const float4* src = (const float4*)(xres + (size_t)row0 * 256);
    float4* dst = (float4*)xt;
#pragma unroll
    for (int i = tid; i < 1024; i += 256) dst[i] = src[i];
  }
  __syncthreads();
  // LN per row (16 threads per row)
  {
    int row = tid >> 4, l16 = tid & 15;
    float s = 0.f, s2 = 0.f;
#pragma unroll
    for (int j = 0; j < 16; ++j) {
      float f = xt[row * 256 + l16 + 16 * j];
      s += f; s2 += f * f;
    }
#pragma unroll
    for (int m = 1; m < 16; m <<= 1) { s += __shfl_xor(s, m, 32); s2 += __shfl_xor(s2, m, 32); }
    float mu = s * (1.f / 256.f);
    float var = s2 * (1.f / 256.f) - mu * mu;
    float ir = rsqrtf(var + EPSF);
#pragma unroll
    for (int j = 0; j < 16; ++j) {
      int c = l16 + 16 * j;
      float f = xt[row * 256 + c];
      a1[row * 256 + c] = f2bf((f - mu) * ir * lnfg[c] + lnfb[c]);
    }
  }
  __syncthreads();

  // GEMM1: (16x256)@(256x2048); fuse bias + a*gelu(g); W1p has 128 column tiles
  for (int p = wave; p < 64; p += 8) {
    v8f ca = {}, cg = {};
#pragma unroll
    for (int kt = 0; kt < 8; ++kt) {
      v16bf a  = lds_a_frag(a1, 0, kt * 32, 256);
      v16bf ba = glb_b_frag_pk(W1p, kt * 128 + p);
      v16bf bg = glb_b_frag_pk(W1p, kt * 128 + 64 + p);
      ca = __builtin_amdgcn_wmma_f32_16x16x32_bf16(false, a, false, ba, (short)0, ca, false, false);
      cg = __builtin_amdgcn_wmma_f32_16x16x32_bf16(false, a, false, bg, (short)0, cg, false, false);
    }
    int nn = lane & 15;
    int mbase = (lane >> 4) << 3;
    float bba = b1[p * 16 + nn];
    float bbg = b1[1024 + p * 16 + nn];
#pragma unroll
    for (int r = 0; r < 8; ++r) {
      float av = ca[r] + bba;
      float gv = cg[r] + bbg;
      float ge = 0.5f * gv * (1.f + erff(gv * 0.70710678118f));   // exact gelu
      gt[(mbase + r) * 1024 + p * 16 + nn] = f2bf(av * ge);
    }
  }
  __syncthreads();

  // GEMM2: (16x1024)@(1024x256) + b2 + residual + relu; W2p has 16 column tiles
  for (int nt = wave; nt < 16; nt += 8) {
    v8f c = {};
#pragma unroll
    for (int kt = 0; kt < 32; ++kt) {
      v16bf a = lds_a_frag(gt, 0, kt * 32, 1024);
      v16bf b = glb_b_frag_pk(W2p, kt * 16 + nt);
      c = __builtin_amdgcn_wmma_f32_16x16x32_bf16(false, a, false, b, (short)0, c, false, false);
    }
    int nn = nt * 16 + (lane & 15);
    int mbase = (lane >> 4) << 3;
#pragma unroll
    for (int r = 0; r < 8; ++r) {
      int m = mbase + r;
      float val = c[r] + b2[nn] + xt[m * 256 + nn];
      out[(size_t)(row0 + m) * 256 + nn] = fmaxf(val, 0.f);
    }
  }
}

// ---------------------------------------------------------------- launch
extern "C" void kernel_launch(void* const* d_in, const int* in_sizes, int n_in,
                              void* d_out, int out_size, void* d_ws, size_t ws_size,
                              hipStream_t stream) {
  (void)in_sizes; (void)n_in; (void)out_size; (void)ws_size;
  const float* image_feats = (const float*)d_in[0];
  const float* point_feats = (const float*)d_in[1];
  const int*   inds2d      = (const int*)d_in[2];
  const int*   inds3d      = (const int*)d_in[3];
  const float* ln_q_g = (const float*)d_in[4];
  const float* ln_q_b = (const float*)d_in[5];
  const float* ln_c_g = (const float*)d_in[6];
  const float* ln_c_b = (const float*)d_in[7];
  const float* Wq = (const float*)d_in[8];
  const float* Wk = (const float*)d_in[9];
  const float* Wv = (const float*)d_in[10];
  const float* Wo = (const float*)d_in[11];
  const float* bo = (const float*)d_in[12];
  const float* ln_f_g = (const float*)d_in[13];
  const float* ln_f_b = (const float*)d_in[14];
  const float* W1 = (const float*)d_in[15];
  const float* b1 = (const float*)d_in[16];
  const float* W2 = (const float*)d_in[17];
  const float* b2 = (const float*)d_in[18];

  const int N = 30000, K = 64, C = 128, M = 1500000;

  // workspace layout
  float* ctx  = (float*)d_ws;                          // N*K*128 f32 (983 MB)
  float* xres = ctx + (size_t)N * K * C;               // N*256 f32 (30.7 MB)
  unsigned int* Wkp = (unsigned int*)(xres + (size_t)N * 256);
  unsigned int* Wvp = Wkp + 8192;                      // 128x128 -> 8192 dwords
  unsigned int* W1p = Wvp + 8192;                      // 256x2048 -> 262144 dwords
  unsigned int* W2p = W1p + 262144;                    // 1024x256 -> 131072 dwords
  float* out = (float*)d_out;

  // pre-pack B weights into bf16 register images (tiny; L2-resident afterwards)
  pack_b_kernel<<<(8192 + 255) / 256, 256, 0, stream>>>(Wk, Wkp, 128, 128);
  pack_b_kernel<<<(8192 + 255) / 256, 256, 0, stream>>>(Wv, Wvp, 128, 128);
  pack_b_kernel<<<(262144 + 255) / 256, 256, 0, stream>>>(W1, W1p, 256, 2048);
  pack_b_kernel<<<(131072 + 255) / 256, 256, 0, stream>>>(W2, W2p, 1024, 256);

  long long n4 = (long long)N * K * C / 4;             // float4 count
  init_ctx_kernel<<<(int)((n4 + 255) / 256), 256, 0, stream>>>((float4*)ctx, n4);

  scatter_kernel<<<M / 8, 256, 0, stream>>>(image_feats, inds2d, inds3d, ctx, M);

  attn_kernel<<<N, 256, 0, stream>>>(point_feats, ctx, ln_q_g, ln_q_b, ln_c_g, ln_c_b,
                                     Wq, Wkp, Wvp, Wo, bo, xres);

  ffn_kernel<<<N / 16, 256, 0, stream>>>(xres, ln_f_g, ln_f_b, W1p, b1, W2p, b2, out);
}
